// mlp_49203145343477
// MI455X (gfx1250) — matrix operand order repaired and checked by reading, not compiled, not run on hardware
//
#include <hip/hip_runtime.h>
#include <hip/hip_bf16.h>

typedef __attribute__((ext_vector_type(16))) _Float16 v16h;
typedef __attribute__((ext_vector_type(8)))  _Float16 v8h;
typedef __attribute__((ext_vector_type(4)))  _Float16 v4h;
typedef __attribute__((ext_vector_type(8)))  float    v8f;
typedef __attribute__((ext_vector_type(4)))  float    v4f;

#define DD 768
#define LL 256
#define BB 32
#define TT 4
#define A_STRIDE 776   // padded LDS row stride in halves: 1552 B -> conflict-free frag reads
#define M_TILE 32      // rows per block (2 WMMA M-tiles per wave)

// ---------------------------------------------------------------------------
// fp32 -> fp16 weight conversion (once per launch; W stays hot in L2 as f16)
// ---------------------------------------------------------------------------
__global__ __launch_bounds__(256) void f2h_kernel(const float* __restrict__ src,
                                                  _Float16* __restrict__ dst, int n) {
    int i = blockIdx.x * 256 + threadIdx.x;
    if (i < n) dst[i] = (_Float16)src[i];
}

// ---------------------------------------------------------------------------
// Fused GEMM (A[rows,768] x W[768,768]^T) + bias + LayerNorm.
// Block = 256 threads (8 waves), 32 x 768 output stripe: each wave owns
// 2 M-tiles x 6 N-tiles -> 12 v_wmma_f32_16x16x32_f16 per K-step.
// All 6 B fragments are loaded into independent buffers before the WMMA
// burst so the scheduler can pipeline loads across K-steps.
// ---------------------------------------------------------------------------
template<bool A_HALF>
__global__ __launch_bounds__(256)
void gemm_ln_kernel(const float* __restrict__ Af32,
                    const _Float16* __restrict__ Af16,
                    const _Float16* __restrict__ W,      // f16 [768 x 768]; row n = W[n, :]
                    const float* __restrict__ bias,
                    const float* __restrict__ gamma,
                    const float* __restrict__ beta,
                    float* __restrict__ out)
{
    // Union: A tile 32 x 776 f16 (49664 B) first, then C tile 32 x 768 f32 (98304 B)
    __shared__ __align__(16) char smem[M_TILE * DD * 4];
    _Float16* As = (_Float16*)smem;
    float*    Cs = (float*)smem;

    const int tid   = threadIdx.x;
    const int lane  = tid & 31;
    const int wave  = tid >> 5;
    const long rowBase = (long)blockIdx.x * M_TILE;

    // ---- cooperative A-tile load: 32 rows x 768, convert to f16 ----
    for (int iv = tid; iv < M_TILE * (DD / 4); iv += 256) {
        int m  = iv / (DD / 4);
        int kv = (iv % (DD / 4)) * 4;
        long src = (rowBase + m) * (long)DD + kv;
        v4h hv;
        if (A_HALF) {
            hv = *(const v4h*)(Af16 + src);
        } else {
            v4f fv = *(const v4f*)(Af32 + src);
            hv = (v4h){(_Float16)fv.x, (_Float16)fv.y, (_Float16)fv.z, (_Float16)fv.w};
        }
        *(v4h*)(As + m * A_STRIDE + kv) = hv;
    }
    __syncthreads();

    // ---- WMMA main loop ----
    v8f acc[2][6];
    v8f vz = {0.f, 0.f, 0.f, 0.f, 0.f, 0.f, 0.f, 0.f};
#pragma unroll
    for (int mt = 0; mt < 2; ++mt)
#pragma unroll
        for (int j = 0; j < 6; ++j) acc[mt][j] = vz;

    const int mrow  = lane & 15;   // A row within an M-tile
    const int halfW = lane >> 4;   // K-half held by this lane
    const int nloc  = lane & 15;   // B column within an N-tile
    const int ncol0 = wave * 96;   // first N column owned by this wave

    // Single base pointer per wave; all B loads become immediate offsets (<128 KB)
    const _Float16* Wbase = W + (long)(ncol0 + nloc) * DD + 16 * halfW;
    const _Float16* A0 = As + mrow * A_STRIDE + 16 * halfW;         // M-tile 0
    const _Float16* A1 = As + (16 + mrow) * A_STRIDE + 16 * halfW;  // M-tile 1

    for (int k0 = 0; k0 < DD; k0 += 32) {
        // Issue all loads for this K-step first: 6 independent B buffers + 2 A frags.
        v16h bf[6];
#pragma unroll
        for (int j = 0; j < 6; ++j)
            bf[j] = *(const v16h*)(Wbase + j * 16 * DD + k0);

        v8h lo0 = *(const v8h*)(A0 + k0);
        v8h hi0 = *(const v8h*)(A0 + k0 + 8);
        v8h lo1 = *(const v8h*)(A1 + k0);
        v8h hi1 = *(const v8h*)(A1 + k0 + 8);
        v16h a0 = __builtin_shufflevector(lo0, hi0, 0,1,2,3,4,5,6,7,8,9,10,11,12,13,14,15);
        v16h a1 = __builtin_shufflevector(lo1, hi1, 0,1,2,3,4,5,6,7,8,9,10,11,12,13,14,15);

        // WMMA burst: 12 matrix ops, each B fragment reused for both M-tiles.
#pragma unroll
        for (int j = 0; j < 6; ++j) {
            acc[0][j] = __builtin_amdgcn_wmma_f32_16x16x32_f16(
                false, a0, false, bf[j], (short)0, acc[0][j], false, false);
            acc[1][j] = __builtin_amdgcn_wmma_f32_16x16x32_f16(
                false, a1, false, bf[j], (short)0, acc[1][j], false, false);
        }
    }
    __syncthreads();   // all waves done reading As before Cs overwrites it

    // ---- spill C + bias to LDS (C layout: vgpr r -> M = r + 8*(lane>=16), N = lane%16) ----
#pragma unroll
    for (int j = 0; j < 6; ++j) {
        int n = ncol0 + j * 16 + nloc;
        float bb = bias[n];
#pragma unroll
        for (int mt = 0; mt < 2; ++mt)
#pragma unroll
            for (int r = 0; r < 8; ++r) {
                int m = mt * 16 + r + 8 * halfW;
                Cs[m * DD + n] = acc[mt][j][r] + bb;
            }
    }
    __syncthreads();

    // ---- fused LayerNorm: wave w normalizes rows 4w .. 4w+3 ----
#pragma unroll
    for (int rr = 0; rr < 4; ++rr) {
        int m = wave * 4 + rr;
        float s = 0.f, ss = 0.f;
        for (int i = lane; i < DD; i += 32) {
            float v = Cs[m * DD + i];
            s += v; ss += v * v;
        }
#pragma unroll
        for (int off = 16; off; off >>= 1) {
            s  += __shfl_xor(s, off, 32);
            ss += __shfl_xor(ss, off, 32);
        }
        const float inv = 1.0f / (float)DD;
        float mu  = s * inv;
        float var = ss * inv - mu * mu;
        float rs  = rsqrtf(var + 1e-5f);
        long ob = (rowBase + m) * (long)DD;
        for (int i = lane; i < DD; i += 32) {
            float v = Cs[m * DD + i];
            out[ob + i] = (v - mu) * rs * gamma[i] + beta[i];
        }
    }
}

// ---------------------------------------------------------------------------
// LIF over t, 4 elements (consecutive d) per thread:
//   v <- (v + x)/2 ; s = (v >= 1) ; v <- v*(1-s)
// Layout [b,t,l,d] -> t-stride = L*D.  Spikes: f16 (layer 1) or f32 (output).
// ---------------------------------------------------------------------------
template<typename OutT>
__global__ __launch_bounds__(256)
void lif_kernel(const float* __restrict__ h, OutT* __restrict__ s)
{
    typedef __attribute__((ext_vector_type(4))) OutT v4o;
    const long LD = (long)LL * DD;                          // 196608
    long flat = ((long)blockIdx.x * 256 + threadIdx.x) * 4; // over B*L*D
    long b = flat / LD;
    long base = flat + b * (TT - 1) * LD;                   // (b, t=0, l, d)
    float v0 = 0.f, v1 = 0.f, v2 = 0.f, v3 = 0.f;
#pragma unroll
    for (int t = 0; t < TT; ++t) {
        v4f x = *(const v4f*)(h + base + (long)t * LD);
        v0 = 0.5f * (v0 + x.x);
        v1 = 0.5f * (v1 + x.y);
        v2 = 0.5f * (v2 + x.z);
        v3 = 0.5f * (v3 + x.w);
        float s0 = (v0 >= 1.0f) ? 1.0f : 0.0f;
        float s1 = (v1 >= 1.0f) ? 1.0f : 0.0f;
        float s2 = (v2 >= 1.0f) ? 1.0f : 0.0f;
        float s3 = (v3 >= 1.0f) ? 1.0f : 0.0f;
        v4o so = {(OutT)s0, (OutT)s1, (OutT)s2, (OutT)s3};
        *(v4o*)(s + base + (long)t * LD) = so;
        v0 *= (1.0f - s0); v1 *= (1.0f - s1);
        v2 *= (1.0f - s2); v3 *= (1.0f - s3);
    }
}

// ---------------------------------------------------------------------------
extern "C" void kernel_launch(void* const* d_in, const int* in_sizes, int n_in,
                              void* d_out, int out_size, void* d_ws, size_t ws_size,
                              hipStream_t stream) {
    (void)in_sizes; (void)n_in; (void)out_size; (void)ws_size;
    const float* x   = (const float*)d_in[0];
    const float* W1  = (const float*)d_in[1];
    const float* b1  = (const float*)d_in[2];
    const float* g1  = (const float*)d_in[3];
    const float* be1 = (const float*)d_in[4];
    const float* W2  = (const float*)d_in[5];
    const float* b2  = (const float*)d_in[6];
    const float* g2  = (const float*)d_in[7];
    const float* be2 = (const float*)d_in[8];
    float* out = (float*)d_out;

    // Workspace carve-up (~153 MB total)
    char* ws = (char*)d_ws;
    const size_t W_H_BYTES = (size_t)DD * DD * 2;           // 1,179,648
    const size_t H_BYTES   = (size_t)BB * TT * LL * DD * 4; // 100,663,296
    _Float16* W1h  = (_Float16*)ws;
    _Float16* W2h  = (_Float16*)(ws + W_H_BYTES);
    float*    hbuf = (float*)   (ws + 2 * W_H_BYTES);       // reused for h2
    _Float16* s1h  = (_Float16*)(ws + 2 * W_H_BYTES + H_BYTES);

    const int WN = DD * DD;                 // 589824
    const int ROWS = BB * TT * LL;          // 32768
    const int ELEMS = BB * LL * DD;         // 6,291,456

    f2h_kernel<<<(WN + 255) / 256, 256, 0, stream>>>(W1, W1h, WN);
    f2h_kernel<<<(WN + 255) / 256, 256, 0, stream>>>(W2, W2h, WN);

    // Layer 1: GEMM + bias + LN (x fp32 -> f16 tiles), then LIF -> f16 spikes
    gemm_ln_kernel<false><<<ROWS / M_TILE, 256, 0, stream>>>(x, nullptr, W1h, b1, g1, be1, hbuf);
    lif_kernel<_Float16><<<ELEMS / (256 * 4), 256, 0, stream>>>(hbuf, s1h);

    // Layer 2: GEMM on binary f16 spikes + bias + LN, then LIF -> f32 output
    gemm_ln_kernel<true><<<ROWS / M_TILE, 256, 0, stream>>>(nullptr, s1h, W2h, b2, g2, be2, hbuf);
    lif_kernel<float><<<ELEMS / (256 * 4), 256, 0, stream>>>(hbuf, out);
}
